// TriangleAttentionStartingNode_43130061586671
// MI455X (gfx1250) — matrix operand order repaired and checked by reading, not compiled, not run on hardware
//
#include <hip/hip_runtime.h>
#include <hip/hip_bf16.h>

typedef _Float16 v16h __attribute__((ext_vector_type(16)));
typedef _Float16 v8h  __attribute__((ext_vector_type(8)));
typedef float    v8f  __attribute__((ext_vector_type(8)));

#define NRES 320
#define NN   (320*320)
#define CCH  128
#define ACH  32
#define HCH  4

// ---------------- WMMA fragment helpers (wave32, 16x16x32 f16) ----------------
// A matrix 16x32 (MxK), row-major source with element stride `strideH`.
// ISA layout: lanes 0-15 row M=lane, v0..3 = K0..7, v4..7 = K16..23;
//             lanes 16-31 row M=lane-16, v0..3 = K8..15, v4..7 = K24..31.
__device__ __forceinline__ v16h load_a_frag(const _Float16* base, int strideH) {
    int lane = threadIdx.x & 31;
    int m = lane & 15, lh = lane >> 4;
    const _Float16* row = base + (long)m * strideH;
    v8h lo = *(const v8h*)(row + lh * 8);
    v8h hi = *(const v8h*)(row + 16 + lh * 8);
    v16h r;
#pragma unroll
    for (int t = 0; t < 8; ++t) { r[t] = lo[t]; r[8 + t] = hi[t]; }
    return r;
}

// B matrix 32x16 (KxN) supplied TRANSPOSED: baseT is [n][k] row-major.
// ISA layout: lanes 0-15 col N=lane hold K0..7,K16..23; lanes 16-31 hold K8..15,K24..31.
__device__ __forceinline__ v16h load_b_frag(const _Float16* baseT, int strideH) {
    int lane = threadIdx.x & 31;
    int n = lane & 15, lh = lane >> 4;
    const _Float16* row = baseT + (long)n * strideH + lh * 8;
    v8h lo = *(const v8h*)(row);
    v8h hi = *(const v8h*)(row + 16);
    v16h r;
#pragma unroll
    for (int t = 0; t < 8; ++t) { r[t] = lo[t]; r[8 + t] = hi[t]; }
    return r;
}

__device__ __forceinline__ v8f wmma_f16(v16h a, v16h b, v8f c) {
    return __builtin_amdgcn_wmma_f32_16x16x32_f16(false, a, false, b, (short)0, c,
                                                  false, false);
}

// ---------------- Kernel 1: weight prep (transpose + f32->f16) ----------------
// wcat rows: [0,128)=Wq^T [128,256)=Wk^T [256,384)=Wv^T [384,512)=Wg^T
//            [512,516)=Wb^T [516,576)=zero pad.  wot = Wo^T.
__global__ void ta_prep_kernel(const float* __restrict__ Wq, const float* __restrict__ Wk,
                               const float* __restrict__ Wv, const float* __restrict__ Wb,
                               const float* __restrict__ Wg, const float* __restrict__ Wo,
                               _Float16* __restrict__ wcat, _Float16* __restrict__ wot) {
    int idx = blockIdx.x * 256 + threadIdx.x;
    if (idx < 576 * 128) {
        int row = idx >> 7, ic = idx & 127;
        float val = 0.f;
        if      (row < 128) val = Wq[ic * 128 + row];
        else if (row < 256) val = Wk[ic * 128 + (row - 128)];
        else if (row < 384) val = Wv[ic * 128 + (row - 256)];
        else if (row < 512) val = Wg[ic * 128 + (row - 384)];
        else if (row < 516) val = Wb[ic * 4 + (row - 512)];
        wcat[idx] = (_Float16)val;
    } else {
        int j = idx - 576 * 128;
        if (j < 128 * 128) {
            int oc = j >> 7, ic = j & 127;
            wot[j] = (_Float16)Wo[ic * 128 + oc];
        }
    }
}

// ---------------- Kernel 2: LayerNorm -> f16 (wave per row) ----------------
__global__ __launch_bounds__(256) void ta_ln_kernel(const float* __restrict__ x,
                                                    const float* __restrict__ w,
                                                    const float* __restrict__ b,
                                                    _Float16* __restrict__ xn) {
    int wave = threadIdx.x >> 5;
    int lane = threadIdx.x & 31;
    long row = (long)blockIdx.x * 8 + wave;  // NN rows total
    const float* xr = x + row * CCH;
    float4 v = *(const float4*)(xr + lane * 4);
    float s  = v.x + v.y + v.z + v.w;
    float ss = v.x * v.x + v.y * v.y + v.z * v.z + v.w * v.w;
#pragma unroll
    for (int off = 16; off > 0; off >>= 1) {
        s  += __shfl_xor(s, off, 32);
        ss += __shfl_xor(ss, off, 32);
    }
    float mu  = s * (1.f / 128.f);
    float var = ss * (1.f / 128.f) - mu * mu;
    float inv = rsqrtf(var + 1e-5f);
    int c0 = lane * 4;
    float vv[4] = {v.x, v.y, v.z, v.w};
#pragma unroll
    for (int t = 0; t < 4; ++t)
        xn[row * CCH + c0 + t] = (_Float16)((vv[t] - mu) * inv * w[c0 + t] + b[c0 + t]);
}

// ---------------- Kernel 3: fused projections (WMMA GEMM, 16x64/wave) ----------
// grid 6400 row-tiles, 288 threads = 9 waves; wave w computes n-tiles 4w..4w+3
// (36 tiles total; tiles >= 33 hit zero-padded weights and are not stored).
__global__ __launch_bounds__(288) void ta_proj_kernel(
    const _Float16* __restrict__ xn, const _Float16* __restrict__ wcat,
    const float* __restrict__ bg,
    _Float16* __restrict__ qb, _Float16* __restrict__ kb, _Float16* __restrict__ vt,
    _Float16* __restrict__ gb, float* __restrict__ bT) {
    int rt = blockIdx.x;
    int wv = threadIdx.x >> 5;        // 0..8
    int lane = threadIdx.x & 31;
    int nn = lane & 15, lh = lane >> 4;
    int nt0 = wv * 4;
    const _Float16* A0 = xn + (long)rt * 16 * CCH;
    __builtin_prefetch(wcat + (long)nt0 * 16 * CCH + lane * 64, 0, 1);
    v8f acc[4] = {};
#pragma unroll
    for (int k0 = 0; k0 < CCH; k0 += 32) {
        v16h a = load_a_frag(A0 + k0, CCH);
#pragma unroll
        for (int t = 0; t < 4; ++t) {
            v16h b = load_b_frag(wcat + (long)(nt0 + t) * 16 * CCH + k0, CCH);
            acc[t] = wmma_f16(a, b, acc[t]);
        }
    }
#pragma unroll
    for (int t = 0; t < 4; ++t) {
        int nt = nt0 + t;
        if (nt > 32) continue;   // wave-uniform
#pragma unroll
        for (int r = 0; r < 8; ++r) {
            int row = rt * 16 + r + 8 * lh;
            int i = row / NRES, j = row % NRES;
            float val = acc[t][r];
            if (nt < 32) {
                int c = (nt * 16 + nn) & 127;   // channel within its projection
                int p = nt >> 3;                // 0=q 1=k 2=v 3=g
                int aI = c >> 2, hI = c & 3;    // c = a*H + h
                if (p == 0)
                    qb[(((long)hI * NRES + i) * NRES + j) * ACH + aI] = (_Float16)val;
                else if (p == 1)
                    kb[(((long)hI * NRES + i) * NRES + j) * ACH + aI] = (_Float16)val;
                else if (p == 2)
                    vt[(((long)hI * NRES + i) * ACH + aI) * NRES + j] = (_Float16)val;
                else {
                    float gv = val + bg[c];
                    gb[(long)row * CCH + c] = (_Float16)(1.f / (1.f + __expf(-gv)));
                }
            } else if (nn < 4) {
                // b projection: braw[i,j,h] stored as bT[h][j][i]
                bT[((long)nn * NRES + j) * NRES + i] = val;
            }
        }
    }
}

// ---------------- Kernel 4: attention, 4 waves/block (one per head) ----------
__global__ __launch_bounds__(128) void ta_attn_kernel(
    const _Float16* __restrict__ qb, const _Float16* __restrict__ kb,
    const _Float16* __restrict__ vt, const _Float16* __restrict__ gb,
    const float* __restrict__ bT, const float* __restrict__ mask,
    _Float16* __restrict__ att) {
    __shared__ __align__(16) _Float16 wp[4][16][NRES];
    int i = blockIdx.x, jt = blockIdx.y;
    int hI = threadIdx.x >> 5;        // wave == head
    int lane = threadIdx.x & 31;
    int nn = lane & 15, lh = lane >> 4;

    const _Float16* qbase = qb + (((long)hI * NRES + i) * NRES + jt * 16) * ACH;
    const _Float16* kbase = kb + ((long)hI * NRES + i) * NRES * ACH;
    const _Float16* vbase = vt + ((long)hI * NRES + i) * ACH * NRES;
    // warm V rows (used after softmax)
    __builtin_prefetch(vbase + lane * NRES, 0, 3);

    v16h afrag = load_a_frag(qbase, ACH);

    // ---- scores: w = QK^T/sqrt(A) + bias, masked; kept register-resident ----
    v8f sacc[20];
#pragma unroll
    for (int kt = 0; kt < 20; ++kt) {
        v16h bfrag = load_b_frag(kbase + (long)kt * 16 * ACH, ACH);
        v8f z = {};
        sacc[kt] = wmma_f16(afrag, bfrag, z);
    }
#pragma unroll
    for (int kt = 0; kt < 20; ++kt) {
        int kk = kt * 16 + nn;
        float mval = mask[i * NRES + kk];
#pragma unroll
        for (int r = 0; r < 8; ++r) {
            int jg = jt * 16 + r + 8 * lh;
            float s = sacc[kt][r] * 0.17677669529663687f +
                      bT[((long)hI * NRES + jg) * NRES + kk];
            sacc[kt][r] = (s + 100.f) * mval - 100.f;
        }
    }

    // ---- softmax over kk: cross-lane tree reduction within 16-lane half ----
    float inv_sum[8];
#pragma unroll
    for (int r = 0; r < 8; ++r) {
        float m = -1e30f;
#pragma unroll
        for (int kt = 0; kt < 20; ++kt) m = fmaxf(m, sacc[kt][r]);
#pragma unroll
        for (int off = 1; off < 16; off <<= 1) m = fmaxf(m, __shfl_xor(m, off, 32));
        float sum = 0.f;
#pragma unroll
        for (int kt = 0; kt < 20; ++kt) {
            float e = __expf(sacc[kt][r] - m);
            sacc[kt][r] = e;
            sum += e;
        }
#pragma unroll
        for (int off = 1; off < 16; off <<= 1) sum += __shfl_xor(sum, off, 32);
        inv_sum[r] = 1.f / sum;
    }

    // transpose probabilities through LDS into A-fragment layout (f16)
#pragma unroll
    for (int kt = 0; kt < 20; ++kt) {
        int kk = kt * 16 + nn;
#pragma unroll
        for (int r = 0; r < 8; ++r)
            wp[hI][r + 8 * lh][kk] = (_Float16)(sacc[kt][r] * inv_sum[r]);
    }

    // ---- out = w @ V, gated ---- (per-wave LDS region; same-wave DS ordering)
#pragma unroll
    for (int ct = 0; ct < 2; ++ct) {
        v8f acc = {};
#pragma unroll
        for (int k2 = 0; k2 < 10; ++k2) {
            v16h a2 = load_a_frag(&wp[hI][0][0] + k2 * 32, NRES);
            v16h b2 = load_b_frag(vbase + (long)(ct * 16) * NRES + k2 * 32, NRES);
            acc = wmma_f16(a2, b2, acc);
        }
#pragma unroll
        for (int r = 0; r < 8; ++r) {
            int jg = jt * 16 + r + 8 * lh;
            int aI = ct * 16 + nn;
            long oidx = ((long)i * NRES + jg) * CCH + aI * HCH + hI;
            att[oidx] = (_Float16)(acc[r] * (float)gb[oidx]);
        }
    }
}

// ---------------- Kernel 5: output projection (16x128 per wave) ----------------
// grid 800 blocks x 8 waves; wave handles one 16-row tile across all 8 n-tiles.
__global__ __launch_bounds__(256) void ta_outproj_kernel(
    const _Float16* __restrict__ att, const _Float16* __restrict__ wot,
    const float* __restrict__ bo, const float* __restrict__ mask,
    float* __restrict__ out) {
    int rt = blockIdx.x * 8 + (threadIdx.x >> 5);
    int lane = threadIdx.x & 31;
    int nn = lane & 15, lh = lane >> 4;
    const _Float16* A0 = att + (long)rt * 16 * CCH;
    __builtin_prefetch(wot + lane * 64, 0, 1);
    v8f acc[8] = {};
#pragma unroll
    for (int k0 = 0; k0 < CCH; k0 += 32) {
        v16h a = load_a_frag(A0 + k0, CCH);
#pragma unroll
        for (int t = 0; t < 8; ++t) {
            v16h b = load_b_frag(wot + (long)t * 16 * CCH + k0, CCH);
            acc[t] = wmma_f16(a, b, acc[t]);
        }
    }
#pragma unroll
    for (int t = 0; t < 8; ++t) {
        int oc = t * 16 + nn;
        float bias = bo[oc];
#pragma unroll
        for (int r = 0; r < 8; ++r) {
            int row = rt * 16 + r + 8 * lh;
            out[(long)row * CCH + oc] = (acc[t][r] + bias) * mask[row];
        }
    }
}

// ---------------- launcher ----------------
extern "C" void kernel_launch(void* const* d_in, const int* in_sizes, int n_in,
                              void* d_out, int out_size, void* d_ws, size_t ws_size,
                              hipStream_t stream) {
    const float* x2d    = (const float*)d_in[0];
    const float* mask   = (const float*)d_in[1];
    const float* norm_w = (const float*)d_in[2];
    const float* norm_b = (const float*)d_in[3];
    const float* Wq     = (const float*)d_in[4];
    const float* Wk     = (const float*)d_in[5];
    const float* Wv     = (const float*)d_in[6];
    const float* Wb     = (const float*)d_in[7];
    const float* Wg     = (const float*)d_in[8];
    const float* bg     = (const float*)d_in[9];
    const float* Wo     = (const float*)d_in[10];
    const float* bo     = (const float*)d_in[11];
    float* out = (float*)d_out;

    char* p = (char*)d_ws;
    _Float16* xnbuf = (_Float16*)p; p += (size_t)NN * CCH * 2;
    _Float16* wcat  = (_Float16*)p; p += (size_t)576 * 128 * 2;
    _Float16* wot   = (_Float16*)p; p += (size_t)128 * 128 * 2;
    _Float16* qbuf  = (_Float16*)p; p += (size_t)HCH * NRES * NRES * ACH * 2;
    _Float16* kbuf  = (_Float16*)p; p += (size_t)HCH * NRES * NRES * ACH * 2;
    _Float16* vtbuf = (_Float16*)p; p += (size_t)HCH * NRES * NRES * ACH * 2;
    _Float16* gbuf  = (_Float16*)p; p += (size_t)NN * CCH * 2;
    float*    bTbuf = (float*)p;    p += (size_t)HCH * NRES * NRES * 4;
    _Float16* attbf = (_Float16*)p; p += (size_t)NN * CCH * 2;

    ta_prep_kernel<<<(576 * 128 + 128 * 128 + 255) / 256, 256, 0, stream>>>(
        Wq, Wk, Wv, Wb, Wg, Wo, wcat, wot);
    ta_ln_kernel<<<NN / 8, 256, 0, stream>>>(x2d, norm_w, norm_b, xnbuf);
    ta_proj_kernel<<<NN / 16, 288, 0, stream>>>(
        xnbuf, wcat, bg, qbuf, kbuf, vtbuf, gbuf, bTbuf);
    ta_attn_kernel<<<dim3(NRES, NRES / 16), 128, 0, stream>>>(
        qbuf, kbuf, vtbuf, gbuf, bTbuf, mask, attbf);
    ta_outproj_kernel<<<NN / (16 * 8), 256, 0, stream>>>(
        attbf, wot, bo, mask, out);
}
